// GraphSAGE_29454885716510
// MI455X (gfx1250) — matrix operand order repaired and checked by reading, not compiled, not run on hardware
//
#include <hip/hip_runtime.h>
#include <hip/hip_bf16.h>

#define N_NODES 100000
#define N_EDGES 1600000
#define D 128

typedef __attribute__((ext_vector_type(16))) __bf16 v16bf;
typedef __attribute__((ext_vector_type(8)))  float  v8f;
typedef __attribute__((__vector_size__(4 * sizeof(int)))) int vi4;   // builtin's pointee type

#define LDSPTR(p)  ((__attribute__((address_space(3))) vi4*)(p))

// round-to-nearest-even f32 -> bf16
__device__ __forceinline__ unsigned short f2bf(float f) {
    unsigned u = __float_as_uint(f);
    u += 0x7FFFu + ((u >> 16) & 1u);
    return (unsigned short)(u >> 16);
}
__device__ __forceinline__ unsigned pack2(float a, float b) {
    return (unsigned)f2bf(a) | ((unsigned)f2bf(b) << 16);
}

__global__ void fill_zero_f32(float* __restrict__ p, int n) {
    int i = blockIdx.x * blockDim.x + threadIdx.x;
    int stride = gridDim.x * blockDim.x;
    for (; i < n; i += stride) p[i] = 0.0f;
}

__global__ void deg_accum(const int* __restrict__ dst, float* __restrict__ deg, int nedges) {
    int e = blockIdx.x * blockDim.x + threadIdx.x;
    if (e < nedges) atomicAdd(&deg[dst[e]], 1.0f);
}

__global__ void deg_invert(float* __restrict__ deg, int n) {
    int i = blockIdx.x * blockDim.x + threadIdx.x;
    if (i < n) deg[i] = 1.0f / fmaxf(deg[i], 1.0f);
}

// 32 lanes per edge, 4 floats per lane (coalesced float4 gather, f32 global atomics
// into an agg buffer that fits in the 192MB L2)
__global__ void scatter_add(const float* __restrict__ xin, const int* __restrict__ src,
                            const int* __restrict__ dst, float* __restrict__ agg, int nedges) {
    int t = blockIdx.x * blockDim.x + threadIdx.x;
    int e = t >> 5;
    if (e >= nedges) return;
    int lane = t & 31;
    int s = src[e], d = dst[e];
    const float4 v = *(const float4*)(xin + (size_t)s * D + lane * 4);
    float* o = agg + (size_t)d * D + lane * 4;
    atomicAdd(o + 0, v.x);
    atomicAdd(o + 1, v.y);
    atomicAdd(o + 2, v.z);
    atomicAdd(o + 3, v.w);
}

// bf16 concat weights, K-major per output column: wcat[layer][n][k 0..255].
// k < 128 -> Wn[k][n] (mean path), k >= 128 -> Ws[k-128][n] (self path).
__global__ void build_wcat(const float* __restrict__ Wn0, const float* __restrict__ Ws0,
                           const float* __restrict__ Wn1, const float* __restrict__ Ws1,
                           const float* __restrict__ Wn2, const float* __restrict__ Ws2,
                           unsigned short* __restrict__ wcat) {
    int idx = blockIdx.x * blockDim.x + threadIdx.x;
    if (idx >= 3 * 128 * 256) return;
    int layer = idx / (128 * 256);
    int rem   = idx % (128 * 256);
    int n = rem / 256;
    int k = rem % 256;
    const float* Wn = (layer == 0) ? Wn0 : (layer == 1) ? Wn1 : Wn2;
    const float* Ws = (layer == 0) ? Ws0 : (layer == 1) ? Ws1 : Ws2;
    float v = (k < 128) ? Wn[k * 128 + n] : Ws[(k - 128) * 128 + n];
    wcat[(size_t)layer * 32768 + (size_t)n * 256 + k] = f2bf(v);
}

// Fused activation prep: acat[node][256] = bf16(concat(agg*deg_inv, x)).
__global__ void prepare_acat(const float* __restrict__ agg, const float* __restrict__ xin,
                             const float* __restrict__ deg_inv,
                             unsigned short* __restrict__ acat) {
    int t = blockIdx.x * blockDim.x + threadIdx.x;
    if (t >= N_NODES * 32) return;
    int node = t >> 5;
    int kc = (t & 31) << 3;           // 0..248
    uint4 o;
    if (kc < D) {
        float dinv = deg_inv[node];
        float4 f0 = *(const float4*)(agg + (size_t)node * D + kc);
        float4 f1 = *(const float4*)(agg + (size_t)node * D + kc + 4);
        o.x = pack2(f0.x * dinv, f0.y * dinv);
        o.y = pack2(f0.z * dinv, f0.w * dinv);
        o.z = pack2(f1.x * dinv, f1.y * dinv);
        o.w = pack2(f1.z * dinv, f1.w * dinv);
    } else {
        float4 f0 = *(const float4*)(xin + (size_t)node * D + kc - D);
        float4 f1 = *(const float4*)(xin + (size_t)node * D + kc - D + 4);
        o.x = pack2(f0.x, f0.y);
        o.y = pack2(f0.z, f0.w);
        o.z = pack2(f1.x, f1.y);
        o.w = pack2(f1.z, f1.w);
    }
    *(uint4*)(acat + (size_t)node * 256 + kc) = o;
}

// One wave -> one 16x16 tile of H = relu?( acat @ wcat + b ), K = 256 in 8 WMMA steps.
// Block = 256 threads = 8 waves = one 16-node slab x all 128 output cols.
// A slab (8KB) and weight block (64KB) staged in LDS (padded: 33-uint4 row stride
// so lane m starts at bank 4m mod 64 -> conflict-free ds_load_b128), via
// GLOBAL_LOAD_ASYNC_TO_LDS_B128 when the builtin is available.
#define WPAD 33                                        // uint4 per 256-k row (padded)
template <int RELU>
__global__ void __launch_bounds__(256) sage_gemm_wmma(
        const unsigned short* __restrict__ acat,
        const unsigned short* __restrict__ wcatT,
        const float* __restrict__ bias, float* __restrict__ out) {
    __shared__ uint4 sW[128 * WPAD];                   // 66KB
    __shared__ uint4 sA[16 * WPAD];                    // 8.25KB
    {
        const uint4* gW = (const uint4*)wcatT;                                    // 4096
        const uint4* gA = (const uint4*)(acat + (size_t)blockIdx.x * 16 * 256);   // 512
#if __has_builtin(__builtin_amdgcn_global_load_async_to_lds_b128)
        #pragma unroll
        for (int i = 0; i < 16; ++i) {
            int idx = threadIdx.x + i * 256;
            __builtin_amdgcn_global_load_async_to_lds_b128(
                (vi4*)(gW + idx), LDSPTR(&sW[(idx >> 5) * WPAD + (idx & 31)]), 0, 0);
        }
        #pragma unroll
        for (int i = 0; i < 2; ++i) {
            int idx = threadIdx.x + i * 256;
            __builtin_amdgcn_global_load_async_to_lds_b128(
                (vi4*)(gA + idx), LDSPTR(&sA[(idx >> 5) * WPAD + (idx & 31)]), 0, 0);
        }
#if __has_builtin(__builtin_amdgcn_s_wait_asynccnt)
        __builtin_amdgcn_s_wait_asynccnt(0);
#else
        asm volatile("s_wait_asynccnt 0" ::: "memory");
#endif
#else
        #pragma unroll
        for (int i = 0; i < 16; ++i) {
            int idx = threadIdx.x + i * 256;
            sW[(idx >> 5) * WPAD + (idx & 31)] = gW[idx];
        }
        #pragma unroll
        for (int i = 0; i < 2; ++i) {
            int idx = threadIdx.x + i * 256;
            sA[(idx >> 5) * WPAD + (idx & 31)] = gA[idx];
        }
#endif
    }
    __syncthreads();

    const int lane = threadIdx.x & 31;
    const int wave = threadIdx.x >> 5;
    const int m = lane & 15;       // A row within tile / B-C-D col within tile
    const int h = lane >> 4;       // lane-half: K sub-stripe (A/B), M-half (C/D)
    const int col0 = wave * 16;

    const uint4* arow = &sA[m * WPAD];                 // 32 data uint4 per row
    const uint4* wcol = &sW[(col0 + m) * WPAD];

    union { v16bf v; uint4 q[2]; } A, B;
    v8f C = {0.f, 0.f, 0.f, 0.f, 0.f, 0.f, 0.f, 0.f};

    #pragma unroll
    for (int k0 = 0; k0 < 256; k0 += 32) {
        // A: half 0 -> K k0..k0+7 and k0+16..23 ; half 1 -> K k0+8..15 and k0+24..31
        const int ai = (k0 >> 3) + h;
        A.q[0] = arow[ai];
        A.q[1] = arow[ai + 2];
        // B: half 0 -> K k0..k0+7 and k0+16..23 ; half 1 -> K k0+8..15 and k0+24..31
        const int bi = (k0 >> 3) + h;
        B.q[0] = wcol[bi];
        B.q[1] = wcol[bi + 2];
        C = __builtin_amdgcn_wmma_f32_16x16x32_bf16(
                /*neg_a=*/false, A.v, /*neg_b=*/false, B.v,
                /*c_mod=*/(short)0, C, /*reuse_a=*/false, /*reuse_b=*/false);
    }

    // D: lanes 0-15 -> M = v, lanes 16-31 -> M = v + 8 ; N = lane&15.
    // Grid is exactly N/16 blocks (N divisible by 16): no row guard needed.
    const float bcol = bias[col0 + m];
    float* orow = out + ((size_t)(blockIdx.x * 16 + h * 8) * D + col0 + m);
    #pragma unroll
    for (int v = 0; v < 8; ++v) {
        float val = C[v] + bcol;
        if (RELU) val = fmaxf(val, 0.0f);
        orow[(size_t)v * D] = val;
    }
}

// y = emb @ Wfc + bfc ; log_softmax over 16 classes. One thread per node.
__global__ void fc_logsoftmax(const float* __restrict__ emb, const float* __restrict__ Wfc,
                              const float* __restrict__ bfc, float* __restrict__ outp) {
    int i = blockIdx.x * blockDim.x + threadIdx.x;
    if (i >= N_NODES) return;
    float y[16];
    #pragma unroll
    for (int j = 0; j < 16; ++j) y[j] = bfc[j];
    const float* er = emb + (size_t)i * D;
    for (int k = 0; k < D; ++k) {
        float e = er[k];
        const float* wr = Wfc + k * 16;
        #pragma unroll
        for (int j = 0; j < 16; ++j) y[j] = fmaf(e, wr[j], y[j]);
    }
    float mx = y[0];
    #pragma unroll
    for (int j = 1; j < 16; ++j) mx = fmaxf(mx, y[j]);
    float s = 0.f;
    #pragma unroll
    for (int j = 0; j < 16; ++j) s += expf(y[j] - mx);
    float lse = logf(s);
    #pragma unroll
    for (int j = 0; j < 16; ++j) outp[(size_t)i * 16 + j] = y[j] - mx - lse;
}

extern "C" void kernel_launch(void* const* d_in, const int* in_sizes, int n_in,
                              void* d_out, int out_size, void* d_ws, size_t ws_size,
                              hipStream_t stream) {
    const float* x   = (const float*)d_in[0];
    const int*   ei  = (const int*)d_in[1];
    const float* Wn0 = (const float*)d_in[2];
    const float* Ws0 = (const float*)d_in[3];
    const float* b0  = (const float*)d_in[4];
    const float* Wn1 = (const float*)d_in[5];
    const float* Ws1 = (const float*)d_in[6];
    const float* b1  = (const float*)d_in[7];
    const float* Wn2 = (const float*)d_in[8];
    const float* Ws2 = (const float*)d_in[9];
    const float* b2  = (const float*)d_in[10];
    const float* Wfc = (const float*)d_in[11];
    const float* bfc = (const float*)d_in[12];
    (void)in_sizes; (void)n_in; (void)out_size; (void)ws_size;

    const int* src = ei;
    const int* dst = ei + N_EDGES;

    float* emb  = (float*)d_out;                          // [N,128] (also h0 temp)
    float* logp = (float*)d_out + (size_t)N_NODES * D;    // [N,16]

    // workspace carve (256B aligned): deg | agg | h1 | acat | wcat  (~154 MB)
    char* w = (char*)d_ws;
    size_t off = 0;
    auto carve = [&](size_t bytes) {
        char* p = w + off;
        off = (off + bytes + 255) & ~(size_t)255;
        return p;
    };
    float* deg = (float*)carve((size_t)N_NODES * 4);
    float* agg = (float*)carve((size_t)N_NODES * D * 4);
    float* h1  = (float*)carve((size_t)N_NODES * D * 4);
    unsigned short* acat = (unsigned short*)carve((size_t)N_NODES * 256 * 2);
    unsigned short* wcat = (unsigned short*)carve((size_t)3 * 256 * 128 * 2);

    const int NAGG = N_NODES * D;
    const int scatterBlocks = (N_EDGES * 32) / 256;       // 200000
    const int gemmBlocks = N_NODES / 16;                  // 6250 (exact)
    const int prepBlocks = (N_NODES * 32 + 255) / 256;    // 12500

    build_wcat<<<(3 * 128 * 256 + 255) / 256, 256, 0, stream>>>(Wn0, Ws0, Wn1, Ws1, Wn2, Ws2, wcat);

    fill_zero_f32<<<1024, 256, 0, stream>>>(deg, N_NODES);
    deg_accum<<<(N_EDGES + 255) / 256, 256, 0, stream>>>(dst, deg, N_EDGES);
    deg_invert<<<(N_NODES + 255) / 256, 256, 0, stream>>>(deg, N_NODES);

    // layer 0: x -> h0 (stored in emb region, dead before the final overwrite)
    fill_zero_f32<<<2048, 256, 0, stream>>>(agg, NAGG);
    scatter_add<<<scatterBlocks, 256, 0, stream>>>(x, src, dst, agg, N_EDGES);
    prepare_acat<<<prepBlocks, 256, 0, stream>>>(agg, x, deg, acat);
    sage_gemm_wmma<1><<<gemmBlocks, 256, 0, stream>>>(acat, wcat, b0, emb);

    // layer 1: h0 -> h1
    fill_zero_f32<<<2048, 256, 0, stream>>>(agg, NAGG);
    scatter_add<<<scatterBlocks, 256, 0, stream>>>(emb, src, dst, agg, N_EDGES);
    prepare_acat<<<prepBlocks, 256, 0, stream>>>(agg, emb, deg, acat);
    sage_gemm_wmma<1><<<gemmBlocks, 256, 0, stream>>>(acat, wcat + 32768, b1, h1);

    // layer 2: h1 -> emb (final, no relu)
    fill_zero_f32<<<2048, 256, 0, stream>>>(agg, NAGG);
    scatter_add<<<scatterBlocks, 256, 0, stream>>>(h1, src, dst, agg, N_EDGES);
    prepare_acat<<<prepBlocks, 256, 0, stream>>>(agg, h1, deg, acat);
    sage_gemm_wmma<0><<<gemmBlocks, 256, 0, stream>>>(acat, wcat + 65536, b2, emb);

    // head
    fc_logsoftmax<<<(N_NODES + 255) / 256, 256, 0, stream>>>(emb, Wfc, bfc, logp);
}
